// MixtureOfExperts_12403865551221
// MI455X (gfx1250) — matrix operand order repaired and checked by reading, not compiled, not run on hardware
//
#include <hip/hip_runtime.h>
#include <stdint.h>

// Problem constants (from reference)
#define TOK   16384      // B*S
#define DIM   1024       // D
#define NEXP  8          // E
#define DFFC  4096       // DFF
#define CAPC  5120       // capacity per expert

#define KSTEP 32         // WMMA K per step (bf16 16x16x32)
#define ROWPAD 40        // shorts per LDS tile row (80B): 16B-aligned chunks, conflict-free

typedef __attribute__((ext_vector_type(16))) __bf16 v16bf;
typedef __attribute__((ext_vector_type(8)))  __bf16 v8bf;
typedef __attribute__((ext_vector_type(8)))  float  v8f;
typedef __attribute__((ext_vector_type(4)))  unsigned vu4;
typedef int v4i __attribute__((vector_size(16)));   // matches async-LDS builtin param type

__device__ __forceinline__ unsigned short f2bf(float f) {
  union { float f; unsigned u; } v; v.f = f;
  unsigned r = v.u + 0x7FFFu + ((v.u >> 16) & 1u);   // round-to-nearest-even
  return (unsigned short)(r >> 16);
}

__device__ __forceinline__ float clip1000(float t) {
  return fminf(fmaxf(t, -1000.f), 1000.f);
}

// ------------------------------------------------------------------
// Global -> LDS staging primitive: async (CDNA5 ASYNCcnt path) when the
// toolchain exposes the builtin, else global_load_b128 + ds_store_b128.
// ------------------------------------------------------------------
#if __has_builtin(__builtin_amdgcn_global_load_async_to_lds_b128)
#define MOE_ASYNC_STAGE 1
__device__ __forceinline__ void copy_b128_g2l(const unsigned short* g, unsigned short* l) {
  __builtin_amdgcn_global_load_async_to_lds_b128(
      (__attribute__((address_space(1))) v4i*)(g),
      (__attribute__((address_space(3))) v4i*)(l), 0, 0);
}
__device__ __forceinline__ void wait_stage() {
#if __has_builtin(__builtin_amdgcn_s_wait_asynccnt)
  __builtin_amdgcn_s_wait_asynccnt(0);
#else
  asm volatile("s_wait_asynccnt 0" ::: "memory");
#endif
}
#else
#define MOE_ASYNC_STAGE 0
__device__ __forceinline__ void copy_b128_g2l(const unsigned short* g, unsigned short* l) {
  *(vu4*)l = *(const vu4*)g;            // global_load_b128 + ds_store_b128
}
__device__ __forceinline__ void wait_stage() {}  // __syncthreads covers DScnt
#endif

// ---------------------------------------------------------------- zero fill
__global__ void moe_zero_u32(unsigned* __restrict__ p, long n) {
  long i = (long)blockIdx.x * blockDim.x + threadIdx.x;
  if (i < n) p[i] = 0u;
}

// ---------------------------------------------------------------- router
// one wave32 per token: coalesced x loads, shfl reduce, lane0 top-2 dispatch
__global__ __launch_bounds__(256) void moe_router(
    const float* __restrict__ x, const float* __restrict__ rw,
    int* __restrict__ counts, int* __restrict__ tokens, float* __restrict__ wts) {
  int t    = (int)((blockIdx.x * blockDim.x + threadIdx.x) >> 5);
  int lane = threadIdx.x & 31;
  if (t >= TOK) return;
  const float* xr = x + (size_t)t * DIM;
  float acc[NEXP];
#pragma unroll
  for (int e = 0; e < NEXP; ++e) acc[e] = 0.f;
  for (int d = lane; d < DIM; d += 32) {
    float xv = xr[d];
    const float* r = rw + (size_t)d * NEXP;
#pragma unroll
    for (int e = 0; e < NEXP; ++e) acc[e] += xv * r[e];
  }
#pragma unroll
  for (int e = 0; e < NEXP; ++e)
#pragma unroll
    for (int off = 16; off > 0; off >>= 1)
      acc[e] += __shfl_xor(acc[e], off, 32);
  if (lane != 0) return;
  float mx = -1e30f;
#pragma unroll
  for (int e = 0; e < NEXP; ++e) {
    acc[e] = fminf(fmaxf(acc[e], -100.f), 100.f);
    mx = fmaxf(mx, acc[e]);
  }
  float p[NEXP], sum = 0.f;
#pragma unroll
  for (int e = 0; e < NEXP; ++e) { p[e] = expf(acc[e] - mx); sum += p[e]; }
#pragma unroll
  for (int e = 0; e < NEXP; ++e) p[e] /= sum;
  int i1 = 0;
#pragma unroll
  for (int e = 1; e < NEXP; ++e) if (p[e] > p[i1]) i1 = e;     // ties -> low idx
  int i2 = (i1 == 0) ? 1 : 0;
#pragma unroll
  for (int e = 0; e < NEXP; ++e) if (e != i1 && p[e] > p[i2]) i2 = e;
  float s = fmaxf(p[i1] + p[i2], 1e-6f);
  float w1 = p[i1] / s, w2 = p[i2] / s;
  int s1 = atomicAdd(&counts[i1], 1);
  if (s1 < CAPC) { tokens[i1 * CAPC + s1] = t; wts[i1 * CAPC + s1] = w1; }
  int s2 = atomicAdd(&counts[i2], 1);
  if (s2 < CAPC) { tokens[i2 * CAPC + s2] = t; wts[i2 * CAPC + s2] = w2; }
}

// ---------------------------------------------------------------- gather xs
// xs[e][slot][d] = bf16(x[token][d]); zero-pad unused capacity slots
__global__ __launch_bounds__(256) void moe_gather(
    const float* __restrict__ x, const int* __restrict__ counts,
    const int* __restrict__ tokens, unsigned short* __restrict__ xs) {
  int row  = blockIdx.x;            // e*CAP + slot
  int e    = row / CAPC;
  int slot = row - e * CAPC;
  unsigned short* dst = xs + (size_t)row * DIM;
  if (slot < counts[e]) {
    const float* src = x + (size_t)tokens[row] * DIM;
    for (int d = threadIdx.x; d < DIM; d += blockDim.x) dst[d] = f2bf(src[d]);
  } else {
    for (int d = threadIdx.x; d < DIM; d += blockDim.x) dst[d] = 0;
  }
}

// ---------------------------------------------------------------- weight transpose+convert
// src f32 [E][R][C]  ->  dst bf16 [E][C][R]   (so B fragments read contiguous K)
__global__ __launch_bounds__(256) void moe_transpose_bf16(
    const float* __restrict__ src, unsigned short* __restrict__ dst, int R, int C) {
  __shared__ float tile[32][33];
  int e = blockIdx.z;
  const float* s     = src + (size_t)e * R * C;
  unsigned short* d  = dst + (size_t)e * R * C;
  int c0 = blockIdx.x * 32, r0 = blockIdx.y * 32;
  int tx = threadIdx.x, ty = threadIdx.y;  // (32, 8)
#pragma unroll
  for (int k = 0; k < 32; k += 8)
    tile[ty + k][tx] = s[(size_t)(r0 + ty + k) * C + (c0 + tx)];
  __syncthreads();
#pragma unroll
  for (int k = 0; k < 32; k += 8)
    d[(size_t)(c0 + ty + k) * R + (r0 + tx)] = f2bf(tile[tx][ty + k]);
}

// ---------------------------------------------------------------- WMMA fragments from LDS
__device__ __forceinline__ v16bf frag16(const unsigned short* p0, const unsigned short* p1) {
  const v8bf lo = *reinterpret_cast<const v8bf*>(p0);
  const v8bf hi = *reinterpret_cast<const v8bf*>(p1);
  return __builtin_shufflevector(lo, hi, 0,1,2,3,4,5,6,7,8,9,10,11,12,13,14,15);
}

// Stage one 128-row x 32-K bf16 tile (16B chunks, 4 per row) into LDS.
__device__ __forceinline__ void stage_tile(
    const unsigned short* __restrict__ gbase, size_t ld, int k0,
    unsigned short* lbase, int tid) {
  const int c = tid & 3;        // 16B chunk in row
  const int r = tid >> 2;       // 0..63
#pragma unroll
  for (int rr = 0; rr < 128; rr += 64) {
    const unsigned short* g = gbase + (size_t)(r + rr) * ld + k0 + c * 8;
    unsigned short* l = lbase + (r + rr) * ROWPAD + c * 8;
    copy_b128_g2l(g, l);
  }
}

// One K-step of WMMA from LDS tiles: wave tile 32(M) x 64(N), 2x4 accumulators.
// A lane(l): M=l&15, K chunks [8*half..+7],[16+8*half..+7]
// B lane(l): N=l&15, K chunks [8*half..+7],[16+8*half..+7]
__device__ __forceinline__ void wmma_step(
    const unsigned short* As, const unsigned short* Bs,
    int wm, int wn, int l16, int half, v8f acc[2][4]) {
  v16bf a[2], b[4];
#pragma unroll
  for (int i = 0; i < 2; ++i) {
    const unsigned short* p = As + (wm * 32 + 16 * i + l16) * ROWPAD;
    a[i] = frag16(p + 8 * half, p + 16 + 8 * half);
  }
#pragma unroll
  for (int j = 0; j < 4; ++j) {
    const unsigned short* p = Bs + (wn * 64 + 16 * j + l16) * ROWPAD;
    b[j] = frag16(p + 8 * half, p + 16 + 8 * half);
  }
#pragma unroll
  for (int i = 0; i < 2; ++i)
#pragma unroll
    for (int j = 0; j < 4; ++j)
      acc[i][j] = __builtin_amdgcn_wmma_f32_16x16x32_bf16(
          false, a[i], false, b[j], (short)0, acc[i][j], false, false);
}

// ---------------------------------------------------------------- GEMM up + GELU
// h[e][m][n] = clip(gelu(xs[e][m][:] . w_upT[e][n][:] + b_up[e][n]))
// 256 thr = 8 waves; block tile 128x128; LDS double-buffered 128x32 A/B tiles
__global__ __launch_bounds__(256) void moe_gemm_up(
    const unsigned short* __restrict__ xs, const unsigned short* __restrict__ wt,
    const float* __restrict__ bias, unsigned short* __restrict__ h) {
  __shared__ unsigned short As[2][128 * ROWPAD];
  __shared__ unsigned short Bs[2][128 * ROWPAD];
  const int e    = blockIdx.z;
  const int tid  = threadIdx.x;
  const int lane = tid & 31;
  const int wave = tid >> 5;
  const int half = lane >> 4;
  const int l16  = lane & 15;
  const int wm   = wave >> 1;           // 0..3
  const int wn   = wave & 1;            // 0..1
  const int m0b  = blockIdx.x * 128;
  const int n0b  = blockIdx.y * 128;
  const unsigned short* Ag = xs + (size_t)e * CAPC * DIM + (size_t)m0b * DIM;
  const unsigned short* Bg = wt + (size_t)e * DFFC * DIM + (size_t)n0b * DIM;

  v8f zero = {};
  v8f acc[2][4];
#pragma unroll
  for (int i = 0; i < 2; ++i)
#pragma unroll
    for (int j = 0; j < 4; ++j) acc[i][j] = zero;

  stage_tile(Ag, DIM, 0, As[0], tid);
  stage_tile(Bg, DIM, 0, Bs[0], tid);
  wait_stage();
  __syncthreads();

  int cur = 0;
  for (int k0 = KSTEP; k0 < DIM; k0 += KSTEP) {
    stage_tile(Ag, DIM, k0, As[cur ^ 1], tid);
    stage_tile(Bg, DIM, k0, Bs[cur ^ 1], tid);
    wmma_step(As[cur], Bs[cur], wm, wn, l16, half, acc);
    wait_stage();
    __syncthreads();
    cur ^= 1;
  }
  wmma_step(As[cur], Bs[cur], wm, wn, l16, half, acc);

  const float* be = bias + (size_t)e * DFFC;
  unsigned short* H = h + (size_t)e * CAPC * DFFC;
  const int m0 = m0b + wm * 32;
  const int n0 = n0b + wn * 64;
#pragma unroll
  for (int j = 0; j < 4; ++j) {
    const int col = n0 + 16 * j + l16;
    const float bb = be[col];
#pragma unroll
    for (int i = 0; i < 2; ++i) {
#pragma unroll
      for (int v = 0; v < 8; ++v) {
        const int row = m0 + 16 * i + v + 8 * half;
        float t = acc[i][j][v] + bb;
        float g = 0.5f * t * (1.f + erff(t * 0.70710678118654752f));  // exact gelu
        H[(size_t)row * DFFC + col] = f2bf(clip1000(g));
      }
    }
  }
}

// ---------------------------------------------------------------- GEMM down + combine
// y[token][n] += clip(h[e][m][:] . w_downT[e][n][:] + b_down[e][n]) * gate[e][m]
__global__ __launch_bounds__(256) void moe_gemm_down(
    const unsigned short* __restrict__ h, const unsigned short* __restrict__ wt,
    const float* __restrict__ bias, const int* __restrict__ tokens,
    const float* __restrict__ wts, float* __restrict__ y) {
  __shared__ unsigned short As[2][128 * ROWPAD];
  __shared__ unsigned short Bs[2][128 * ROWPAD];
  const int e    = blockIdx.z;
  const int tid  = threadIdx.x;
  const int lane = tid & 31;
  const int wave = tid >> 5;
  const int half = lane >> 4;
  const int l16  = lane & 15;
  const int wm   = wave >> 1;
  const int wn   = wave & 1;
  const int m0b  = blockIdx.x * 128;
  const int n0b  = blockIdx.y * 128;
  const unsigned short* Ag = h  + (size_t)e * CAPC * DFFC + (size_t)m0b * DFFC;
  const unsigned short* Bg = wt + (size_t)e * DIM * DFFC + (size_t)n0b * DFFC;

  v8f zero = {};
  v8f acc[2][4];
#pragma unroll
  for (int i = 0; i < 2; ++i)
#pragma unroll
    for (int j = 0; j < 4; ++j) acc[i][j] = zero;

  stage_tile(Ag, DFFC, 0, As[0], tid);
  stage_tile(Bg, DFFC, 0, Bs[0], tid);
  wait_stage();
  __syncthreads();

  int cur = 0;
  for (int k0 = KSTEP; k0 < DFFC; k0 += KSTEP) {
    stage_tile(Ag, DFFC, k0, As[cur ^ 1], tid);
    stage_tile(Bg, DFFC, k0, Bs[cur ^ 1], tid);
    wmma_step(As[cur], Bs[cur], wm, wn, l16, half, acc);
    wait_stage();
    __syncthreads();
    cur ^= 1;
  }
  wmma_step(As[cur], Bs[cur], wm, wn, l16, half, acc);

  const float* be = bias + (size_t)e * DIM;
  const int m0 = m0b + wm * 32;
  const int n0 = n0b + wn * 64;
#pragma unroll
  for (int i = 0; i < 2; ++i) {
    int   tok[8];
    float g[8];
#pragma unroll
    for (int v = 0; v < 8; ++v) {
      const int slot = m0 + 16 * i + v + 8 * half;
      tok[v] = tokens[e * CAPC + slot];
      g[v]   = wts[e * CAPC + slot];    // 0 for padded slots -> contributes 0
    }
#pragma unroll
    for (int j = 0; j < 4; ++j) {
      const int col = n0 + 16 * j + l16;
      const float bb = be[col];
#pragma unroll
      for (int v = 0; v < 8; ++v) {
        float t = clip1000(acc[i][j][v] + bb) * g[v];
        __hip_atomic_fetch_add(&y[(size_t)tok[v] * DIM + col], t,
                               __ATOMIC_RELAXED, __HIP_MEMORY_SCOPE_AGENT);
      }
    }
  }
}

// ---------------------------------------------------------------- host launcher
extern "C" void kernel_launch(void* const* d_in, const int* in_sizes, int n_in,
                              void* d_out, int out_size, void* d_ws, size_t ws_size,
                              hipStream_t stream) {
  const float* x      = (const float*)d_in[0];
  const float* rw     = (const float*)d_in[1];
  const float* w_up   = (const float*)d_in[2];
  const float* b_up   = (const float*)d_in[3];
  const float* w_down = (const float*)d_in[4];
  const float* b_down = (const float*)d_in[5];
  float* y = (float*)d_out;

  // workspace carve-up (256B aligned slabs)
  char* ws = (char*)d_ws;
  size_t off = 0;
  auto carve = [&](size_t bytes) -> void* {
    void* p = ws + off;
    off = (off + bytes + 255) & ~(size_t)255;
    return p;
  };
  int*            counts = (int*)carve(64 * sizeof(int));              // padded to 256B
  int*            tokens = (int*)carve((size_t)NEXP * CAPC * 4);
  float*          gate   = (float*)carve((size_t)NEXP * CAPC * 4);
  unsigned short* xs     = (unsigned short*)carve((size_t)NEXP * CAPC * DIM * 2);
  unsigned short* w_upT  = (unsigned short*)carve((size_t)NEXP * DFFC * DIM * 2);
  unsigned short* w_dnT  = (unsigned short*)carve((size_t)NEXP * DIM * DFFC * 2);
  unsigned short* hbuf   = (unsigned short*)carve((size_t)NEXP * CAPC * DFFC * 2);

  // 1) zero metadata (counts+tokens+gate are contiguous) and output
  {
    long nmeta = 64 + (long)NEXP * CAPC * 2;
    moe_zero_u32<<<(unsigned)((nmeta + 255) / 256), 256, 0, stream>>>((unsigned*)counts, nmeta);
    long ny = (long)TOK * DIM;
    moe_zero_u32<<<(unsigned)((ny + 255) / 256), 256, 0, stream>>>((unsigned*)y, ny);
  }
  // 2) router: one wave per token
  moe_router<<<(TOK * 32) / 256, 256, 0, stream>>>(x, rw, counts, tokens, gate);
  // 3) gather activations into per-expert bf16 matrices
  moe_gather<<<NEXP * CAPC, 256, 0, stream>>>(x, counts, tokens, xs);
  // 4) weight transpose+convert: w_up [E][D][DFF] -> [E][DFF][D], w_down [E][DFF][D] -> [E][D][DFF]
  moe_transpose_bf16<<<dim3(DFFC / 32, DIM / 32, NEXP), dim3(32, 8), 0, stream>>>(w_up, w_upT, DIM, DFFC);
  moe_transpose_bf16<<<dim3(DIM / 32, DFFC / 32, NEXP), dim3(32, 8), 0, stream>>>(w_down, w_dnT, DFFC, DIM);
  // 5) expert MLP, WMMA bf16, LDS double-buffered async staging
  moe_gemm_up<<<dim3(CAPC / 128, DFFC / 128, NEXP), 256, 0, stream>>>(xs, w_upT, b_up, hbuf);
  moe_gemm_down<<<dim3(CAPC / 128, DIM / 128, NEXP), 256, 0, stream>>>(hbuf, w_dnT, b_down, tokens, gate, y);
}
